// instance_CL_loss_88905823027973
// MI455X (gfx1250) — compile-verified, matrix operand order checked by reading, not run on hardware
//
#include <hip/hip_runtime.h>

#define NROWS 4096
#define D 512
#define TWO_N 8192
#define M_BLK 128     // rows per workgroup (8 waves x 16)
#define CHUNK 512     // columns per workgroup
#define NTILE 8       // 16-wide column tiles per group (N=128)
#define NCHUNK (TWO_N / CHUNK)   // 16
#define KP 32         // K-panel per async stage (one WMMA k-step)
#define NGROUP (CHUNK / (NTILE * 16))          // 4
#define NSTAGE (NGROUP * (D / KP))             // 4 * 16 = 64
#define BROW 144      // LDS bytes per staged column: hi 64 + lo 64 + 16 pad (bank-friendly)

typedef __attribute__((ext_vector_type(16))) __bf16 v16bf;
typedef __attribute__((ext_vector_type(8)))  float  v8f;
typedef __attribute__((ext_vector_type(4)))  unsigned int v4u;
typedef int v4i __attribute__((vector_size(16)));   // generic vector type for async builtin

union Frag { v16bf b; v4u u[2]; };

// ---- workspace layout (bytes) ----
#define HI_OFF   0ul
#define LO_OFF   (HI_OFF + (size_t)TWO_N * D * 2)          // 8 MiB
#define PART_OFF (LO_OFF + (size_t)TWO_N * D * 2)          // 16 MiB
#define LOSS_OFF (PART_OFF + (size_t)NCHUNK * TWO_N * 4)   // +512 KiB

// ---- gfx1250 async-to-LDS path (feature-detected, safe fallback) ----
#if defined(__has_builtin)
# if __has_builtin(__builtin_amdgcn_global_load_async_to_lds_b128) && \
     __has_builtin(__builtin_amdgcn_s_wait_asynccnt)
#  define USE_ASYNC 1
# endif
#endif
#ifndef USE_ASYNC
# define USE_ASYNC 0
#endif

static __device__ __forceinline__ void ldg2lds16(const unsigned short* src, char* dst) {
#if USE_ASYNC
    __builtin_amdgcn_global_load_async_to_lds_b128(
        (__attribute__((address_space(1))) v4i*)(void*)src,
        (__attribute__((address_space(3))) v4i*)(void*)dst, 0, 0);
#else
    *(v4u*)dst = *(const v4u*)src;
#endif
}
static __device__ __forceinline__ void async_wait_all() {
#if USE_ASYNC
    __builtin_amdgcn_s_wait_asynccnt(0);
#endif
}

static __device__ __forceinline__ unsigned short f2bf_rne(float x) {
    unsigned u = __float_as_uint(x);
    unsigned r = u + 0x7FFFu + ((u >> 16) & 1u);
    return (unsigned short)(r >> 16);
}
static __device__ __forceinline__ float bf2f(unsigned short h) {
    return __uint_as_float(((unsigned)h) << 16);
}

// ---------------- K1: f32 -> bf16 hi/lo planes ----------------
__global__ __launch_bounds__(256) void cl_convert_kernel(
    const float* __restrict__ f1, const float* __restrict__ f2,
    unsigned short* __restrict__ hi, unsigned short* __restrict__ lo) {
    long t = (long)blockIdx.x * blockDim.x + threadIdx.x;
    long e = t * 4;
    int row = (int)(e / D);
    int col = (int)(e % D);
    const float* src = (row < NROWS) ? (f1 + (long)row * D)
                                     : (f2 + (long)(row - NROWS) * D);
    float4 x = *(const float4*)(src + col);
    ushort4 h, l;
    h.x = f2bf_rne(x.x); l.x = f2bf_rne(x.x - bf2f(h.x));
    h.y = f2bf_rne(x.y); l.y = f2bf_rne(x.y - bf2f(h.y));
    h.z = f2bf_rne(x.z); l.z = f2bf_rne(x.z - bf2f(h.z));
    h.w = f2bf_rne(x.w); l.w = f2bf_rne(x.w - bf2f(h.w));
    *(ushort4*)(hi + e) = h;
    *(ushort4*)(lo + e) = l;
}

// ---------------- K2: WMMA Gram tile + fused exp-rowsum ----------------
__global__ __launch_bounds__(256) void cl_gram_exp_kernel(
    const unsigned short* __restrict__ hi, const unsigned short* __restrict__ lo,
    float* __restrict__ partial) {
    __shared__ __align__(16) char smem[2][128 * BROW];  // 2 x 18432 B, double buffered

    const int tid   = threadIdx.x;
    const int lane  = tid & 31;
    const int wave  = tid >> 5;               // 0..7
    const int l16   = lane & 15;
    const int halfk = (lane >> 4) * 8;        // ISA half-wave k sub-offset
    const int r0    = blockIdx.x * M_BLK + wave * 16;
    const int c0    = blockIdx.y * CHUNK;

    const unsigned a_base = (unsigned)(r0 + l16) * D + halfk;
    const v8f vzero = {0.f, 0.f, 0.f, 0.f, 0.f, 0.f, 0.f, 0.f};

    // --- precomputed per-thread staging geometry ---
    // stage = 128 cols x 32 k x (hi+lo) = 1024 x 16B chunks; 4 chunks/thread.
    // chunk ci: col = ci>>3, sub = ci&7 (0..3 hi plane, 4..7 lo plane)
    const int ssub   = tid & 7;                          // same for all 4 chunks
    const int scol0  = tid >> 3;                         // + j*32
    const unsigned short* splane = (ssub & 4) ? lo : hi;
    unsigned stg_goff[4];                                // per-thread global elem offsets (minus uniform part)
    unsigned stg_doff[4];                                // per-thread LDS byte offsets
#pragma unroll
    for (int j = 0; j < 4; ++j) {
        const int col = scol0 + j * 32;
        stg_goff[j] = (unsigned)col * D + (ssub & 3) * 8;
        stg_doff[j] = (unsigned)col * BROW + ssub * 16;
    }

    float racc[8];
#pragma unroll
    for (int v = 0; v < 8; ++v) racc[v] = 0.0f;
    v8f c[NTILE];

    // stage st: group = st>>4, k-panel = (st&15)*KP
    {
        const unsigned uni0 = (unsigned)c0 * D;          // gbase=c0, kp=0
#pragma unroll
        for (int j = 0; j < 4; ++j)
            ldg2lds16(splane + uni0 + stg_goff[j], smem[0] + stg_doff[j]);
    }

    for (int st = 0; st < NSTAGE; ++st) {
        async_wait_all();
        __syncthreads();                      // buffer st&1 ready for all waves
        if (st + 1 < NSTAGE) {                // overlap next DMA with compute
            const int nst = st + 1;
            const unsigned uni = (unsigned)(c0 + (nst >> 4) * (NTILE * 16)) * D
                               + (unsigned)((nst & 15) * KP);
            char* nbuf = smem[nst & 1];
#pragma unroll
            for (int j = 0; j < 4; ++j)
                ldg2lds16(splane + uni + stg_goff[j], nbuf + stg_doff[j]);
        }

        const char* sbuf = smem[st & 1];
        if ((st & 15) == 0) {
#pragma unroll
            for (int t = 0; t < NTILE; ++t) c[t] = vzero;
        }
        const int k0 = (st & 15) * KP;

        // A fragment (16 rows x 32 k) from global, hi + lo
        Frag ah, al;
        const unsigned short* pah = hi + (a_base + k0);
        const unsigned short* pal = lo + (a_base + k0);
        ah.u[0] = *(const v4u*)(pah);
        ah.u[1] = *(const v4u*)(pah + 16);
        al.u[0] = *(const v4u*)(pal);
        al.u[1] = *(const v4u*)(pal + 16);

#pragma unroll
        for (int t = 0; t < NTILE; ++t) {
            const char* rb = sbuf + (unsigned)(t * 16 + l16) * BROW
                           + (unsigned)halfk * 2;
            Frag bh, bl;
            bh.u[0] = *(const v4u*)(rb);
            bh.u[1] = *(const v4u*)(rb + 32);
            bl.u[0] = *(const v4u*)(rb + 64);
            bl.u[1] = *(const v4u*)(rb + 96);
            // bf16x3 fp32 emulation: hi*hi + hi*lo + lo*hi
            c[t] = __builtin_amdgcn_wmma_f32_16x16x32_bf16(
                false, ah.b, false, bh.b, (short)0, c[t], false, false);
            c[t] = __builtin_amdgcn_wmma_f32_16x16x32_bf16(
                false, ah.b, false, bl.b, (short)0, c[t], false, false);
            c[t] = __builtin_amdgcn_wmma_f32_16x16x32_bf16(
                false, al.b, false, bh.b, (short)0, c[t], false, false);
        }

        if ((st & 15) == 15) {                // end of a 128-column group: fused exp
#pragma unroll
            for (int t = 0; t < NTILE; ++t)
#pragma unroll
                for (int v = 0; v < 8; ++v)
                    racc[v] += __expf(2.0f * c[t][v]);
        }
    }

    // reduce across each 16-lane half (C layout: vgpr v -> rows v / v+8)
#pragma unroll
    for (int v = 0; v < 8; ++v) {
        float s = racc[v];
        s += __shfl_xor(s, 1, 32);
        s += __shfl_xor(s, 2, 32);
        s += __shfl_xor(s, 4, 32);
        s += __shfl_xor(s, 8, 32);
        racc[v] = s;
    }
    if (l16 == 0) {
        int rbase = r0 + ((lane >> 4) ? 8 : 0);
        float* dst = partial + (long)blockIdx.y * TWO_N + rbase;
#pragma unroll
        for (int v = 0; v < 8; ++v) dst[v] = racc[v];
    }
}

// ---------------- K3: per-row diag/pair dots (exact f32) + loss term ----------------
__global__ __launch_bounds__(256) void cl_rowloss_kernel(
    const float* __restrict__ f1, const float* __restrict__ f2,
    const float* __restrict__ partial, float* __restrict__ loss) {
    const int lane = threadIdx.x & 31;
    const int wave = threadIdx.x >> 5;
    const int row  = blockIdx.x * 8 + wave;
    const float* xi = (row < NROWS) ? (f1 + (long)row * D)
                                    : (f2 + (long)(row - NROWS) * D);
    const int p = (row < NROWS) ? (row + NROWS) : (row - NROWS);
    const float* xp = (p < NROWS) ? (f1 + (long)p * D)
                                  : (f2 + (long)(p - NROWS) * D);
    float dii = 0.0f, dip = 0.0f;
#pragma unroll 4
    for (int j = lane; j < D; j += 32) {
        float a = xi[j];
        dii += a * a;
        dip += a * xp[j];
    }
#pragma unroll
    for (int m = 1; m < 32; m <<= 1) {
        dii += __shfl_xor(dii, m, 32);
        dip += __shfl_xor(dip, m, 32);
    }
    if (lane == 0) {
        float rs = 0.0f;
#pragma unroll
        for (int cidx = 0; cidx < NCHUNK; ++cidx)
            rs += partial[(long)cidx * TWO_N + row];
        float num = __expf(2.0f * dip);
        float den = rs - __expf(2.0f * dii) - num;
        loss[row] = -8.0f * __logf(num / den);
    }
}

// ---------------- K4: mean over 8192 loss terms ----------------
__global__ __launch_bounds__(256) void cl_mean_kernel(
    const float* __restrict__ loss, float* __restrict__ out) {
    __shared__ float smem[256];
    float s = 0.0f;
#pragma unroll
    for (int j = 0; j < TWO_N / 256; ++j)
        s += loss[threadIdx.x + 256 * j];
    smem[threadIdx.x] = s;
    __syncthreads();
    for (int off = 128; off > 0; off >>= 1) {
        if ((int)threadIdx.x < off) smem[threadIdx.x] += smem[threadIdx.x + off];
        __syncthreads();
    }
    if (threadIdx.x == 0) out[0] = smem[0] * (1.0f / (float)TWO_N);
}

extern "C" void kernel_launch(void* const* d_in, const int* in_sizes, int n_in,
                              void* d_out, int out_size, void* d_ws, size_t ws_size,
                              hipStream_t stream) {
    const float* f1 = (const float*)d_in[0];
    const float* f2 = (const float*)d_in[1];
    char* ws = (char*)d_ws;
    unsigned short* hi = (unsigned short*)(ws + HI_OFF);
    unsigned short* lo = (unsigned short*)(ws + LO_OFF);
    float* partial     = (float*)(ws + PART_OFF);
    float* loss        = (float*)(ws + LOSS_OFF);
    float* out         = (float*)d_out;

    cl_convert_kernel<<<dim3((TWO_N * D / 4) / 256), dim3(256), 0, stream>>>(f1, f2, hi, lo);
    cl_gram_exp_kernel<<<dim3(TWO_N / M_BLK, NCHUNK), dim3(256), 0, stream>>>(hi, lo, partial);
    cl_rowloss_kernel<<<dim3(TWO_N / 8), dim3(256), 0, stream>>>(f1, f2, partial, loss);
    cl_mean_kernel<<<dim3(1), dim3(256), 0, stream>>>(loss, out);
}
